// CostModel_76759655514335
// MI455X (gfx1250) — matrix operand order repaired and checked by reading, not compiled, not run on hardware
//
#include <hip/hip_runtime.h>
#include <hip/hip_bf16.h>
#include <math.h>

#define HID 128
#define HEADS 4
#define HD 32
#define CFG_DIM 16
#define HW_DIM 8
#define G_DIM 16
#define BGRAPH 64

typedef __attribute__((ext_vector_type(16))) _Float16 v16h;
typedef __attribute__((ext_vector_type(8)))  _Float16 v8h;
typedef __attribute__((ext_vector_type(8)))  float    v8f;

static __device__ __forceinline__ v16h cat8(v8h lo, v8h hi) {
  return __builtin_shufflevector(lo, hi, 0,1,2,3,4,5,6,7,8,9,10,11,12,13,14,15);
}
static __device__ __forceinline__ v8f wmma_f16(v16h a, v16h b, v8f c) {
  // D = A(16x32 f16) * B(32x16 f16) + C(16x16 f32)
  return __builtin_amdgcn_wmma_f32_16x16x32_f16(false, a, false, b, (short)0, c, false, false);
}

// ---------------------------------------------------------------------------
// Stage 0: X = relu(concat(x, s[batch], h[batch]) @ Win + bin)   [N,56]->[N,128]
// K padded 56 -> 64 (2 WMMA k-steps). One wave = 16 rows x 128 cols.
// ---------------------------------------------------------------------------
__global__ __launch_bounds__(256) void k_input_gemm(
    const float* __restrict__ x, const float* __restrict__ s,
    const float* __restrict__ h, const int* __restrict__ batch,
    const float* __restrict__ Win, const float* __restrict__ bin,
    float* __restrict__ X, int n) {
  const int LSTR = 72;                       // 64 + 8 pad halfs (16B-aligned stride)
  __shared__ __align__(16) _Float16 Wt[128 * 72];   // Wt[col*LSTR + k] = Win[k][col]
  int tid = threadIdx.x;
  for (int idx = tid; idx < 56 * 128; idx += 256) {
    int k = idx >> 7, c = idx & 127;
    Wt[c * LSTR + k] = (_Float16)Win[idx];
  }
  for (int idx = tid; idx < 8 * 128; idx += 256) {  // zero-pad k = 56..63
    int k = 56 + (idx >> 7), c = idx & 127;
    Wt[c * LSTR + k] = (_Float16)0.f;
  }
  __syncthreads();

  int wave = tid >> 5, lane = tid & 31;
  int m = lane & 15, hi = lane >> 4;
  int row0 = (blockIdx.x * 8 + wave) * 16;
  bool full = (row0 + 16) <= n;              // uniform within wave
  int row = row0 + m;
  int rc = row < n ? row : n - 1;
  int b = batch[rc];

  v16h afr[2];
  #pragma unroll
  for (int ks = 0; ks < 2; ks++) {
    #pragma unroll
    for (int i = 0; i < 16; i++) {
      int kl = ks * 32 + (i < 8 ? hi * 8 + i : 16 + hi * 8 + (i - 8));
      float v;
      if (kl < 32)      v = x[(size_t)rc * 32 + kl];
      else if (kl < 48) v = s[b * CFG_DIM + (kl - 32)];
      else if (kl < 56) v = h[b * HW_DIM + (kl - 48)];
      else              v = 0.f;
      afr[ks][i] = (_Float16)v;
    }
  }
  #pragma unroll
  for (int nt = 0; nt < 8; nt++) {
    int col = nt * 16 + m;
    v8f c = {};
    #pragma unroll
    for (int ks = 0; ks < 2; ks++) {
      const _Float16* q = &Wt[col * LSTR + ks * 32 + hi * 8];
      c = wmma_f16(afr[ks], cat8(*(const v8h*)q, *(const v8h*)(q + 16)), c);
    }
    float bb = bin[col];
    float* outp = X + (size_t)(row0 + 8 * hi) * HID + col;
    if (full) {
      #pragma unroll
      for (int r = 0; r < 8; r++) {
        float v = c[r] + bb;
        outp[(size_t)r * HID] = v > 0.f ? v : 0.f;
      }
    } else {
      #pragma unroll
      for (int r = 0; r < 8; r++) {
        if (row0 + r + 8 * hi < n) {
          float v = c[r] + bb;
          outp[(size_t)r * HID] = v > 0.f ? v : 0.f;
        }
      }
    }
  }
}

// ---------------------------------------------------------------------------
// xp = X @ W   [N,128]@[128,128]  (no bias, no activation)
// ---------------------------------------------------------------------------
__global__ __launch_bounds__(256) void k_gemm128(
    const float* __restrict__ A, const float* __restrict__ W,
    float* __restrict__ Out, int n) {
  const int LSTR = 136;                      // 128 + 8 pad halfs
  __shared__ __align__(16) _Float16 Wt[128 * 136];  // Wt[col*LSTR + k] = W[k][col]
  int tid = threadIdx.x;
  for (int idx = tid; idx < 128 * 128; idx += 256) {
    int k = idx >> 7, c = idx & 127;
    Wt[c * LSTR + k] = (_Float16)W[idx];
  }
  __syncthreads();

  int wave = tid >> 5, lane = tid & 31;
  int m = lane & 15, hi = lane >> 4;
  int row0 = (blockIdx.x * 8 + wave) * 16;
  bool full = (row0 + 16) <= n;              // uniform within wave
  int row = row0 + m;
  size_t rc = (size_t)(full ? row : (row < n ? row : n - 1));

  v16h afr[4];
  #pragma unroll
  for (int ks = 0; ks < 4; ks++) {
    const float* p = A + rc * HID + ks * 32 + hi * 8;
    #pragma unroll
    for (int i = 0; i < 8; i++) {
      afr[ks][i]     = (_Float16)p[i];
      afr[ks][i + 8] = (_Float16)p[16 + i];
    }
  }
  #pragma unroll
  for (int nt = 0; nt < 8; nt++) {
    int col = nt * 16 + m;
    v8f c = {};
    #pragma unroll
    for (int ks = 0; ks < 4; ks++) {
      const _Float16* q = &Wt[col * LSTR + ks * 32 + hi * 8];
      c = wmma_f16(afr[ks], cat8(*(const v8h*)q, *(const v8h*)(q + 16)), c);
    }
    float* outp = Out + (size_t)(row0 + 8 * hi) * HID + col;
    if (full) {
      #pragma unroll
      for (int r = 0; r < 8; r++) outp[(size_t)r * HID] = c[r];
    } else {
      #pragma unroll
      for (int r = 0; r < 8; r++)
        if (row0 + r + 8 * hi < n) outp[(size_t)r * HID] = c[r];
    }
  }
}

// ---------------------------------------------------------------------------
// a_src[n,h] = <xp[n,h,:], att_src[h,:]>, a_dst likewise.  4 waves/node, 1 head/wave.
// ---------------------------------------------------------------------------
__global__ __launch_bounds__(128) void k_attn_coef(
    const float* __restrict__ xp, const float* __restrict__ as_,
    const float* __restrict__ ad_, float* __restrict__ a_src,
    float* __restrict__ a_dst) {
  int node = blockIdx.x;
  int hh = threadIdx.x >> 5, d = threadIdx.x & 31;
  float v = xp[(size_t)node * HID + hh * HD + d];
  float vs = v * as_[hh * HD + d];
  float vd = v * ad_[hh * HD + d];
  #pragma unroll
  for (int o = 16; o > 0; o >>= 1) {
    vs += __shfl_down(vs, o);
    vd += __shfl_down(vd, o);
  }
  if (d == 0) { a_src[node * HEADS + hh] = vs; a_dst[node * HEADS + hh] = vd; }
}

__global__ void k_fill(float* __restrict__ p, float v, int nw) {
  int t = blockIdx.x * blockDim.x + threadIdx.x;
  if (t < nw) p[t] = v;
}

static __device__ __forceinline__ void atomicMaxF(float* addr, float val) {
  if (val >= 0.f) atomicMax((int*)addr, __float_as_int(val));
  else            atomicMin((unsigned int*)addr, __float_as_uint(val));
}

// edges e<E come from edge_index; e>=E are self-loops (src=dst=e-E)
__global__ void k_edge_max(const int* __restrict__ ei, int E_, int n,
                           const float* __restrict__ a_src, const float* __restrict__ a_dst,
                           float* __restrict__ amax) {
  int e = blockIdx.x * blockDim.x + threadIdx.x;
  int total = E_ + n;
  if (e >= total) return;
  int s, d;
  if (e < E_) { s = ei[e]; d = ei[E_ + e]; } else { s = e - E_; d = s; }
  float4 av = *(const float4*)(a_src + (size_t)s * 4);
  float4 bv = *(const float4*)(a_dst + (size_t)d * 4);
  float al[4] = {av.x + bv.x, av.y + bv.y, av.z + bv.z, av.w + bv.w};
  #pragma unroll
  for (int hh = 0; hh < 4; hh++) {
    float a = al[hh] >= 0.f ? al[hh] : 0.2f * al[hh];
    atomicMaxF(&amax[(size_t)d * 4 + hh], a);
  }
}

__global__ void k_edge_denom(const int* __restrict__ ei, int E_, int n,
                             const float* __restrict__ a_src, const float* __restrict__ a_dst,
                             const float* __restrict__ amax, float* __restrict__ denom) {
  int e = blockIdx.x * blockDim.x + threadIdx.x;
  int total = E_ + n;
  if (e >= total) return;
  int s, d;
  if (e < E_) { s = ei[e]; d = ei[E_ + e]; } else { s = e - E_; d = s; }
  float4 av = *(const float4*)(a_src + (size_t)s * 4);
  float4 bv = *(const float4*)(a_dst + (size_t)d * 4);
  float4 mv = *(const float4*)(amax + (size_t)d * 4);
  float al[4] = {av.x + bv.x, av.y + bv.y, av.z + bv.z, av.w + bv.w};
  float mm[4] = {mv.x, mv.y, mv.z, mv.w};
  #pragma unroll
  for (int hh = 0; hh < 4; hh++) {
    float a = al[hh] >= 0.f ? al[hh] : 0.2f * al[hh];
    float m = mm[hh]; if (!(m >= -3.0e38f)) m = 0.f;   // isfinite clamp
    atomicAdd(&denom[(size_t)d * 4 + hh], __expf(a - m));
  }
}

// one wave per edge: lane = feature d, loop over 4 heads; recompute attn
__global__ __launch_bounds__(256) void k_edge_scatter(
    const int* __restrict__ ei, int E_, int n,
    const float* __restrict__ a_src, const float* __restrict__ a_dst,
    const float* __restrict__ amax, const float* __restrict__ denom,
    const float* __restrict__ xp, float* __restrict__ acc) {
  int gw = (blockIdx.x * blockDim.x + threadIdx.x) >> 5;
  int lane = threadIdx.x & 31;
  int total = E_ + n;
  if (gw >= total) return;
  int s, d;
  if (gw < E_) { s = ei[gw]; d = ei[E_ + gw]; } else { s = gw - E_; d = s; }
  float4 av = *(const float4*)(a_src + (size_t)s * 4);
  float4 bv = *(const float4*)(a_dst + (size_t)d * 4);
  float4 mv = *(const float4*)(amax + (size_t)d * 4);
  float4 dv = *(const float4*)(denom + (size_t)d * 4);
  float aa[4] = {av.x + bv.x, av.y + bv.y, av.z + bv.z, av.w + bv.w};
  float mm[4] = {mv.x, mv.y, mv.z, mv.w};
  float dd[4] = {dv.x, dv.y, dv.z, dv.w};
  float attn[4];
  #pragma unroll
  for (int hh = 0; hh < 4; hh++) {
    float a = aa[hh] >= 0.f ? aa[hh] : 0.2f * aa[hh];
    float m = mm[hh]; if (!(m >= -3.0e38f)) m = 0.f;
    attn[hh] = __expf(a - m) / (dd[hh] + 1e-16f);
  }
  const float* xs = xp + (size_t)s * HID;
  float* od = acc + (size_t)d * HID;
  #pragma unroll
  for (int hh = 0; hh < 4; hh++) {
    int c = hh * HD + lane;
    atomicAdd(&od[c], xs[c] * attn[hh]);
  }
}

__global__ void k_bias_relu(float* __restrict__ X, const float* __restrict__ b, int nelem) {
  int t = blockIdx.x * blockDim.x + threadIdx.x;
  if (t < nelem) {
    float v = X[t] + b[t & 127];
    X[t] = v > 0.f ? v : 0.f;
  }
}

__global__ void k_pool(const float* __restrict__ X, const int* __restrict__ batch,
                       float* __restrict__ pooled, int nelem) {
  int t = blockIdx.x * blockDim.x + threadIdx.x;
  if (t < nelem) {
    int node = t >> 7, c = t & 127;
    atomicAdd(&pooled[batch[node] * HID + c], X[t]);
  }
}

// ------------------------------ MLP head (B=64, tiny) ------------------------
__global__ __launch_bounds__(256) void k_head1(
    const float* __restrict__ pooled, const float* __restrict__ s,
    const float* __restrict__ h, const float* __restrict__ g,
    const float* __restrict__ W, const float* __restrict__ bias,
    float* __restrict__ z1) {
  int b = blockIdx.x, j = threadIdx.x;
  float acc = bias[j];
  for (int k = 0; k < 168; k++) {
    float cv;
    if (k < 128)      cv = pooled[b * 128 + k];
    else if (k < 144) cv = s[b * 16 + (k - 128)];
    else if (k < 152) cv = h[b * 8 + (k - 144)];
    else              cv = g[b * 16 + (k - 152)];
    acc += cv * W[k * 256 + j];
  }
  z1[b * 256 + j] = acc > 0.f ? acc : 0.f;
}

__global__ __launch_bounds__(128) void k_head2(
    const float* __restrict__ z1, const float* __restrict__ W,
    const float* __restrict__ bias, float* __restrict__ z2) {
  int b = blockIdx.x, j = threadIdx.x;
  float acc = bias[j];
  for (int k = 0; k < 256; k++) acc += z1[b * 256 + k] * W[k * 128 + j];
  z2[b * 128 + j] = acc > 0.f ? acc : 0.f;
}

__global__ __launch_bounds__(64) void k_head3(
    const float* __restrict__ z2, const float* __restrict__ W,
    const float* __restrict__ bias, float* __restrict__ out) {
  int b = threadIdx.x;
  if (b < 64) {
    float acc = bias[0];
    for (int k = 0; k < 128; k++) acc += z2[b * 128 + k] * W[k];
    out[b] = acc;
  }
}

// ---------------------------------------------------------------------------
extern "C" void kernel_launch(void* const* d_in, const int* in_sizes, int n_in,
                              void* d_out, int out_size, void* d_ws, size_t ws_size,
                              hipStream_t stream) {
  const float* x   = (const float*)d_in[0];
  const float* s   = (const float*)d_in[1];
  const float* h   = (const float*)d_in[2];
  const float* g   = (const float*)d_in[3];
  const int*   ei  = (const int*)d_in[4];
  const int*   bat = (const int*)d_in[5];
  const float* Win = (const float*)d_in[6];
  const float* bin = (const float*)d_in[7];
  const float* W0  = (const float*)d_in[8];
  const float* as0 = (const float*)d_in[9];
  const float* ad0 = (const float*)d_in[10];
  const float* b0  = (const float*)d_in[11];
  const float* W1  = (const float*)d_in[12];
  const float* as1 = (const float*)d_in[13];
  const float* ad1 = (const float*)d_in[14];
  const float* b1  = (const float*)d_in[15];
  const float* Wh1 = (const float*)d_in[16];
  const float* bh1 = (const float*)d_in[17];
  const float* Wh2 = (const float*)d_in[18];
  const float* bh2 = (const float*)d_in[19];
  const float* Wh3 = (const float*)d_in[20];
  const float* bh3 = (const float*)d_in[21];
  float* out = (float*)d_out;
  (void)n_in; (void)out_size; (void)ws_size;

  const int N = in_sizes[5];        // nodes
  const int E = in_sizes[4] / 2;    // edges (edge_index is [2,E])

  char* ws = (char*)d_ws;
  size_t off = 0;
  auto take = [&](size_t bytes) {
    char* p = ws + off;
    off += (bytes + 255) & ~(size_t)255;
    return p;
  };
  float* X      = (float*)take((size_t)N * HID * 4);       // xh / message accumulator
  float* P      = (float*)take((size_t)N * HID * 4);       // xp
  float* a_src  = (float*)take((size_t)N * HEADS * 4);
  float* a_dst  = (float*)take((size_t)N * HEADS * 4);
  float* amax   = (float*)take((size_t)N * HEADS * 4);
  float* denom  = (float*)take((size_t)N * HEADS * 4);
  float* pooled = (float*)take((size_t)BGRAPH * HID * 4);
  float* z1     = (float*)take((size_t)BGRAPH * 256 * 4);
  float* z2     = (float*)take((size_t)BGRAPH * HID * 4);

  int rowBlocks = (N + 127) / 128;
  int total = E + N;
  int nelem = N * HID;

  k_input_gemm<<<rowBlocks, 256, 0, stream>>>(x, s, h, bat, Win, bin, X, N);

  const float* Ws[2]  = {W0, W1};
  const float* ass[2] = {as0, as1};
  const float* ads[2] = {ad0, ad1};
  const float* bs[2]  = {b0, b1};
  for (int layer = 0; layer < 2; layer++) {
    k_gemm128<<<rowBlocks, 256, 0, stream>>>(X, Ws[layer], P, N);
    k_attn_coef<<<N, 128, 0, stream>>>(P, ass[layer], ads[layer], a_src, a_dst);
    k_fill<<<(N * HEADS + 255) / 256, 256, 0, stream>>>(amax, -INFINITY, N * HEADS);
    k_fill<<<(N * HEADS + 255) / 256, 256, 0, stream>>>(denom, 0.f, N * HEADS);
    k_fill<<<(nelem + 255) / 256, 256, 0, stream>>>(X, 0.f, nelem);
    k_edge_max<<<(total + 255) / 256, 256, 0, stream>>>(ei, E, N, a_src, a_dst, amax);
    k_edge_denom<<<(total + 255) / 256, 256, 0, stream>>>(ei, E, N, a_src, a_dst, amax, denom);
    k_edge_scatter<<<(total * 32 + 255) / 256, 256, 0, stream>>>(ei, E, N, a_src, a_dst,
                                                                 amax, denom, P, X);
    k_bias_relu<<<(nelem + 255) / 256, 256, 0, stream>>>(X, bs[layer], nelem);
  }

  k_fill<<<(BGRAPH * HID + 255) / 256, 256, 0, stream>>>(pooled, 0.f, BGRAPH * HID);
  k_pool<<<(nelem + 255) / 256, 256, 0, stream>>>(X, bat, pooled, nelem);
  k_head1<<<BGRAPH, 256, 0, stream>>>(pooled, s, h, g, Wh1, bh1, z1);
  k_head2<<<BGRAPH, 128, 0, stream>>>(z1, Wh2, bh2, z2);
  k_head3<<<1, 64, 0, stream>>>(z2, Wh3, bh3, out);
}
